// GATGraph_25314537242715
// MI455X (gfx1250) — matrix operand order repaired and checked by reading, not compiled, not run on hardware
//
#include <hip/hip_runtime.h>
#include <hip/hip_bf16.h>

typedef __attribute__((ext_vector_type(16))) _Float16 v16h;
typedef __attribute__((ext_vector_type(8)))  _Float16 v8h;
typedef __attribute__((ext_vector_type(8)))  float    v8f;

struct H16 { v8h lo, hi; };

#define NEG_SLOPE 0.2f
#define ENC_NEG_INF 0x007FFFFFu

// Monotonic encode of float into unsigned so atomicMax works for any sign.
__device__ __forceinline__ unsigned encF(float f) {
    unsigned u = __float_as_uint(f);
    return (u & 0x80000000u) ? ~u : (u | 0x80000000u);
}
__device__ __forceinline__ float decF(unsigned u) {
    u = (u & 0x80000000u) ? (u & 0x7FFFFFFFu) : ~u;
    return __uint_as_float(u);
}

// ---------------------------------------------------------------------------
// f32 -> f16 elementwise convert (A operand, row-major preserved).
// ---------------------------------------------------------------------------
__global__ void f32_to_f16(const float* __restrict__ in, _Float16* __restrict__ out,
                           int n) {
    const int i = blockIdx.x * blockDim.x + threadIdx.x;
    if (i < n) out[i] = (_Float16)in[i];
}

// ---------------------------------------------------------------------------
// Weights: B[K x NcReal] f32 -> BT[Nc x K] f16 (transposed, zero-padded cols).
// ---------------------------------------------------------------------------
__global__ void w_to_f16_t(const float* __restrict__ B, _Float16* __restrict__ BT,
                           int K, int NcReal, int Nc) {
    const int idx = blockIdx.x * blockDim.x + threadIdx.x;
    if (idx >= Nc * K) return;
    const int n = idx / K, k = idx - n * K;
    BT[idx] = (n < NcReal) ? (_Float16)B[(size_t)k * NcReal + n] : (_Float16)0.0f;
}

// ---------------------------------------------------------------------------
// WMMA GEMM: C[M x Nc] = A[M x K] * B  with A row-major f16 and BT = B^T
// row-major f16 ([Nc x K], pre-padded). One wave per 16x16 tile; per K-step
// of 32: 4 unconditional b128 loads + 1 v_wmma_f32_16x16x32_f16.
// Requires M % 16 == 0, Nc % 16 == 0, K % 32 == 0.
//
// A fragment (16-bit A 16x32): halves 0..7  <- K = k0 + hi*8 + (0..7)
//                              halves 8..15 <- K = k0 + 16 + hi*8 + (0..7)
// B fragment (16-bit B 32x16): lane col n = lane&15, same K order as A fragment
// C/D: VGPR r -> row = r + hi*8, col = lane&15.
// ---------------------------------------------------------------------------
__global__ void __launch_bounds__(32)
wmma_gemm_f16(const _Float16* __restrict__ A,
              const _Float16* __restrict__ BT,
              float* __restrict__ C,
              int M, int K, int Nc) {
    const int lane = threadIdx.x;       // 0..31
    const int lo   = lane & 15;
    const int hi   = lane >> 4;

    const _Float16* arow = A  + (size_t)(blockIdx.x * 16 + lo) * K;
    const _Float16* brow = BT + (size_t)(blockIdx.y * 16 + lo) * K;

    v8f acc = {};
    for (int k0 = 0; k0 < K; k0 += 32) {
        H16 at, bt;
        at.lo = *(const v8h*)(arow + k0 + hi * 8);
        at.hi = *(const v8h*)(arow + k0 + 16 + hi * 8);
        bt.lo = *(const v8h*)(brow + k0 + hi * 8);
        bt.hi = *(const v8h*)(brow + k0 + 16 + hi * 8);
        const v16h av = __builtin_bit_cast(v16h, at);
        const v16h bv = __builtin_bit_cast(v16h, bt);
        acc = __builtin_amdgcn_wmma_f32_16x16x32_f16(
            false, av, false, bv, (short)0, acc, false, false);
    }

    const int mBase = blockIdx.x * 16 + hi * 8;
    const int n     = blockIdx.y * 16 + lo;
#pragma unroll
    for (int r = 0; r < 8; ++r)
        C[(size_t)(mBase + r) * Nc + n] = acc[r];
}

// ---------------------------------------------------------------------------
// Per (node, head): a_src/a_dst dot products + init amax/denom.
// ---------------------------------------------------------------------------
__global__ void attn_prep(const float* __restrict__ xh,
                          const float* __restrict__ atts,
                          const float* __restrict__ attd,
                          float* __restrict__ as_, float* __restrict__ ad_,
                          unsigned* __restrict__ amax, float* __restrict__ den,
                          int N, int H, int C, int rowStride) {
    const int idx = blockIdx.x * blockDim.x + threadIdx.x;
    if (idx >= N * H) return;
    const int n = idx / H, h = idx - n * H;
    const float* row = xh + (size_t)n * rowStride + h * C;
    float ss = 0.f, sd = 0.f;
    for (int c = 0; c < C; ++c) {
        const float v = row[c];
        ss += v * atts[h * C + c];
        sd += v * attd[h * C + c];
    }
    as_[idx] = ss;
    ad_[idx] = sd;
    amax[idx] = ENC_NEG_INF;
    den[idx]  = 0.f;
}

// Edge pass 1: alpha = leaky_relu(a_src[s]+a_dst[d]); segment max via atomicMax.
__global__ void edge_score(const int* __restrict__ src, const int* __restrict__ dst,
                           const float* __restrict__ as_, const float* __restrict__ ad_,
                           unsigned* __restrict__ amax, float* __restrict__ alphaBuf,
                           int E, int ET, int H) {
    const int idx = blockIdx.x * blockDim.x + threadIdx.x;
    if (idx >= ET * H) return;
    const int e = idx / H, h = idx - e * H;
    const int s = (e < E) ? src[e] : (e - E);
    const int d = (e < E) ? dst[e] : (e - E);
    float a = as_[s * H + h] + ad_[d * H + h];
    a = (a > 0.f) ? a : NEG_SLOPE * a;
    alphaBuf[idx] = a;
    atomicMax(&amax[d * H + h], encF(a));
}

// Edge pass 2: e = exp(alpha - amax[dst]); denom[dst] += e.
__global__ void edge_exp(const int* __restrict__ src, const int* __restrict__ dst,
                         const unsigned* __restrict__ amax, float* __restrict__ alphaBuf,
                         float* __restrict__ den, int E, int ET, int H) {
    const int idx = blockIdx.x * blockDim.x + threadIdx.x;
    if (idx >= ET * H) return;
    const int e = idx / H, h = idx - e * H;
    const int d = (e < E) ? dst[e] : (e - E);
    const float ev = expf(alphaBuf[idx] - decF(amax[d * H + h]));
    alphaBuf[idx] = ev;
    atomicAdd(&den[d * H + h], ev);
}

// Layer-1 aggregation: one 256-thread block per edge (H=8, C=32 -> 256 lanes),
// coalesced gather of xh1[src] row, coalesced atomic scatter to out[dst].
__global__ void edge_agg_l1(const int* __restrict__ src, const int* __restrict__ dst,
                            const float* __restrict__ xh1, const float* __restrict__ ebuf,
                            const float* __restrict__ den, float* __restrict__ out,
                            int E) {
    const int e = blockIdx.x;
    const int t = threadIdx.x;          // 0..255 ; h = t>>5
    const int h = t >> 5;
    const int s = (e < E) ? src[e] : (e - E);
    const int d = (e < E) ? dst[e] : (e - E);
    const float w = ebuf[e * 8 + h] / den[d * 8 + h];
    atomicAdd(&out[(size_t)d * 256 + t], xh1[(size_t)s * 256 + t] * w);
}

// Layer-2 aggregation: thread per (edge, padded channel), H=1, C_out=10.
__global__ void edge_agg_l2(const int* __restrict__ src, const int* __restrict__ dst,
                            const float* __restrict__ xh2, const float* __restrict__ ebuf,
                            const float* __restrict__ den, float* __restrict__ out,
                            int E, int ET) {
    const int idx = blockIdx.x * blockDim.x + threadIdx.x;
    const int e = idx >> 4, c = idx & 15;
    if (e >= ET || c >= 10) return;
    const int s = (e < E) ? src[e] : (e - E);
    const int d = (e < E) ? dst[e] : (e - E);
    const float w = ebuf[e] / den[d];
    atomicAdd(&out[(size_t)d * 16 + c], xh2[(size_t)s * 16 + c] * w);
}

// Layer-1 epilogue: h16[n,c] = (f16) elu(in[n,c] + bias[c]); feeds GEMM2.
__global__ void elu_bias_f16(const float* __restrict__ in, const float* __restrict__ bias,
                             _Float16* __restrict__ out16, int total, int Cpad) {
    const int idx = blockIdx.x * blockDim.x + threadIdx.x;
    if (idx >= total) return;
    const int c = idx % Cpad;
    const float v = in[idx] + bias[c];
    out16[idx] = (_Float16)((v > 0.f) ? v : expm1f(v));
}

// Layer-2 epilogue: buf[n,c] = elu(buf[n,c] + bias[c]) in place (padded stride).
__global__ void elu_bias(float* __restrict__ buf, const float* __restrict__ bias,
                         int total, int Cpad, int Creal) {
    const int idx = blockIdx.x * blockDim.x + threadIdx.x;
    if (idx >= total) return;
    const int c = idx % Cpad;
    if (c < Creal) {
        const float v = buf[idx] + bias[c];
        buf[idx] = (v > 0.f) ? v : expm1f(v);
    }
}

// Global mean pool accumulation (padded h2 stride 16, 10 real channels).
__global__ void pool_accum(const float* __restrict__ h2, const int* __restrict__ batch,
                           float* __restrict__ pooled, float* __restrict__ cnt, int N) {
    const int n = blockIdx.x * blockDim.x + threadIdx.x;
    if (n >= N) return;
    const int g = batch[n];
#pragma unroll
    for (int c = 0; c < 10; ++c)
        atomicAdd(&pooled[g * 10 + c], h2[(size_t)n * 16 + c]);
    atomicAdd(&cnt[g], 1.0f);
}

// mean + log_softmax per graph; one thread per graph.
__global__ void final_logsoftmax(const float* __restrict__ pooled,
                                 const float* __restrict__ cnt,
                                 float* __restrict__ out, int G) {
    const int g = blockIdx.x * blockDim.x + threadIdx.x;
    if (g >= G) return;
    const float inv = 1.0f / fmaxf(cnt[g], 1.0f);
    float v[10], m = -3.4e38f;
#pragma unroll
    for (int c = 0; c < 10; ++c) { v[c] = pooled[g * 10 + c] * inv; m = fmaxf(m, v[c]); }
    float s = 0.f;
#pragma unroll
    for (int c = 0; c < 10; ++c) s += expf(v[c] - m);
    const float lse = logf(s);
#pragma unroll
    for (int c = 0; c < 10; ++c) out[g * 10 + c] = v[c] - m - lse;
}

extern "C" void kernel_launch(void* const* d_in, const int* in_sizes, int n_in,
                              void* d_out, int out_size, void* d_ws, size_t ws_size,
                              hipStream_t stream) {
    const float* x      = (const float*)d_in[0];
    const int*   eidx   = (const int*)d_in[1];
    const int*   batch  = (const int*)d_in[2];
    const float* W1     = (const float*)d_in[3];
    const float* atts1  = (const float*)d_in[4];
    const float* attd1  = (const float*)d_in[5];
    const float* b1     = (const float*)d_in[6];
    const float* W2     = (const float*)d_in[7];
    const float* atts2  = (const float*)d_in[8];
    const float* attd2  = (const float*)d_in[9];
    const float* b2     = (const float*)d_in[10];
    float* out = (float*)d_out;

    const int F = 128, H = 8, C = 32, HC = 256, C2 = 10, C2P = 16, G = 128;
    const int N  = in_sizes[0] / F;          // 50000 (multiple of 16)
    const int E  = in_sizes[1] / 2;          // 800000
    const int ET = E + N;                    // with self-loops
    const int* src = eidx;
    const int* dst = eidx + E;

    // -------- workspace layout --------
    float* ws = (float*)d_ws;
    float*    xh1   = ws;                              // N*HC f32
    float*    out1  = xh1  + (size_t)N * HC;           // N*HC f32 (agg buffer)
    float*    as1   = out1 + (size_t)N * HC;           // N*H
    float*    ad1   = as1  + (size_t)N * H;
    unsigned* amax1 = (unsigned*)(ad1 + (size_t)N * H);
    float*    den1  = (float*)(amax1 + (size_t)N * H);
    float*    e1    = den1 + (size_t)N * H;            // ET*H
    // f16 operand staging region
    _Float16* x16   = (_Float16*)(e1 + (size_t)ET * H);// N*F
    _Float16* w1t   = x16  + (size_t)N * F;            // HC*F
    _Float16* h1f16 = w1t  + (size_t)HC * F;           // N*HC
    _Float16* w2t   = h1f16 + (size_t)N * HC;          // C2P*HC
    // layer-2 buffers reuse the xh1 region (free after agg_l1 / elu)
    float*    xh2   = xh1;                             // N*16
    float*    as2   = xh2  + (size_t)N * C2P;          // N
    float*    ad2   = as2  + N;
    unsigned* amax2 = (unsigned*)(ad2 + N);
    float*    den2  = (float*)(amax2 + N);
    float*    e2    = den2 + N;                        // ET
    float*    out2  = e2   + ET;                       // N*16
    float*    pooled= out2 + (size_t)N * C2P;          // G*10
    float*    cnt   = pooled + G * C2;                 // G

    // ================= Layer 1 =================
    // operand conversion (f16 staging)
    f32_to_f16<<<((size_t)N * F + 255) / 256, 256, 0, stream>>>(x, x16, N * F);
    w_to_f16_t<<<(HC * F + 255) / 256, 256, 0, stream>>>(W1, w1t, F, HC, HC);
    // GEMM: xh1 = x @ W1 (WMMA f16, f32 acc)
    {
        dim3 grid(N / 16, HC / 16);
        wmma_gemm_f16<<<grid, 32, 0, stream>>>(x16, w1t, xh1, N, F, HC);
    }
    // attention scores + state init
    attn_prep<<<(N * H + 255) / 256, 256, 0, stream>>>(
        xh1, atts1, attd1, as1, ad1, amax1, den1, N, H, C, HC);
    // edge softmax
    edge_score<<<((size_t)ET * H + 255) / 256, 256, 0, stream>>>(
        src, dst, as1, ad1, amax1, e1, E, ET, H);
    edge_exp<<<((size_t)ET * H + 255) / 256, 256, 0, stream>>>(
        src, dst, amax1, e1, den1, E, ET, H);
    // aggregation
    hipMemsetAsync(out1, 0, (size_t)N * HC * sizeof(float), stream);
    edge_agg_l1<<<ET, 256, 0, stream>>>(src, dst, xh1, e1, den1, out1, E);
    // h1 = elu(out1 + b1) -> f16 operand for GEMM2
    elu_bias_f16<<<((size_t)N * HC + 255) / 256, 256, 0, stream>>>(
        out1, b1, h1f16, N * HC, HC);

    // ================= Layer 2 =================
    w_to_f16_t<<<(C2P * HC + 255) / 256, 256, 0, stream>>>(W2, w2t, HC, C2, C2P);
    {
        dim3 grid(N / 16, C2P / 16);
        wmma_gemm_f16<<<grid, 32, 0, stream>>>(h1f16, w2t, xh2, N, HC, C2P);
    }
    attn_prep<<<(N + 255) / 256, 256, 0, stream>>>(
        xh2, atts2, attd2, as2, ad2, amax2, den2, N, 1, C2, C2P);
    edge_score<<<(ET + 255) / 256, 256, 0, stream>>>(
        src, dst, as2, ad2, amax2, e2, E, ET, 1);
    edge_exp<<<(ET + 255) / 256, 256, 0, stream>>>(
        src, dst, amax2, e2, den2, E, ET, 1);
    hipMemsetAsync(out2, 0, (size_t)N * C2P * sizeof(float), stream);
    edge_agg_l2<<<((size_t)ET * C2P + 255) / 256, 256, 0, stream>>>(
        src, dst, xh2, e2, den2, out2, E, ET);
    elu_bias<<<((size_t)N * C2P + 255) / 256, 256, 0, stream>>>(
        out2, b2, N * C2P, C2P, C2);

    // ================= Pool + log_softmax =================
    hipMemsetAsync(pooled, 0, (size_t)(G * C2 + G) * sizeof(float), stream);
    pool_accum<<<(N + 255) / 256, 256, 0, stream>>>(out2, batch, pooled, cnt, N);
    final_logsoftmax<<<1, 128, 0, stream>>>(pooled, cnt, out, G);
}
